// BatchMatMul_46282567581944
// MI455X (gfx1250) — 8-bit matrix operand order repaired and checked by reading, compiled, not run on hardware
//
#include <hip/hip_runtime.h>

typedef __attribute__((ext_vector_type(8))) int   v8i;
typedef __attribute__((ext_vector_type(4))) int   v4i;

#define B_DIM   16
#define MATDIM  1024
#define QSCALE  10.0f
#define OSCALE_INV (1.0f / 10.0f)

#define BLK_M 128
#define BLK_N 128
#define BK    64
#define KSTEPS (MATDIM / BK)

#define TILE_BYTES (BLK_M * BK)
#define CHUNK1_G   (64 * MATDIM)
#define CHUNK1_L   (64 * BK)

__device__ __forceinline__ int quant1(float x) {
    float r = __builtin_rintf(x * QSCALE);
    r = fminf(fmaxf(r, -128.0f), 127.0f);
    return (int)r;
}

__device__ __forceinline__ void async_copy_b128(unsigned ldsOff, const signed char* gsrc, int imm) {
    asm volatile("global_load_async_to_lds_b128 %0, %1, off offset:%2"
                 :: "v"(ldsOff), "v"(gsrc), "i"(imm)
                 : "memory");
}

__device__ __forceinline__ void async_wait0() {
    asm volatile("s_wait_asynccnt 0x0" ::: "memory");
}

__global__ __launch_bounds__(256) void quantA_kernel(const float* __restrict__ in,
                                                     signed char* __restrict__ out) {
    const size_t i = (size_t)blockIdx.x * blockDim.x + threadIdx.x;
    const float4 v = ((const float4*)in)[i];
    const unsigned b0 = (unsigned)quant1(v.x) & 0xffu;
    const unsigned b1 = (unsigned)quant1(v.y) & 0xffu;
    const unsigned b2 = (unsigned)quant1(v.z) & 0xffu;
    const unsigned b3 = (unsigned)quant1(v.w) & 0xffu;
    ((unsigned*)out)[i] = b0 | (b1 << 8) | (b2 << 16) | (b3 << 24);
}

__global__ __launch_bounds__(256) void quantBT_kernel(const float* __restrict__ in,
                                                      signed char* __restrict__ out) {
    __shared__ signed char tile[32][36];
    const int tid = threadIdx.x;
    const size_t base = (size_t)blockIdx.z * MATDIM * MATDIM;
    const int k0 = blockIdx.y * 32;
    const int n0 = blockIdx.x * 32;

    const float* src = in + base + (size_t)k0 * MATDIM + n0;
#pragma unroll
    for (int i = 0; i < 4; ++i) {
        const int idx = tid + 256 * i;
        const int kk  = idx >> 5;
        const int nn  = idx & 31;
        tile[kk][nn] = (signed char)quant1(src[(size_t)kk * MATDIM + nn]);
    }
    __syncthreads();

    const int nLocal = tid >> 3;
    const int kq     = (tid & 7) * 4;
    const unsigned pack = ((unsigned)(unsigned char)tile[kq + 0][nLocal])
                        | ((unsigned)(unsigned char)tile[kq + 1][nLocal] << 8)
                        | ((unsigned)(unsigned char)tile[kq + 2][nLocal] << 16)
                        | ((unsigned)(unsigned char)tile[kq + 3][nLocal] << 24);
    signed char* dst = out + base + (size_t)(n0 + nLocal) * MATDIM + k0;
    *(unsigned*)(dst + kq) = pack;
}

__global__ __launch_bounds__(256) void gemm_i8_wmma_kernel(const signed char* __restrict__ qa,
                                                           const signed char* __restrict__ qbt,
                                                           float* __restrict__ out) {
    __shared__ signed char lA[2][TILE_BYTES];
    __shared__ signed char lB[2][TILE_BYTES];

    const int tid  = threadIdx.x;
    const int lane = tid & 31;
    const int wave = tid >> 5;
    const int wm   = wave >> 1;
    const int wn   = wave & 1;
    const int row  = lane & 15;
    const int hi   = lane >> 4;

    const size_t MK    = (size_t)MATDIM * MATDIM;
    const size_t batch = (size_t)blockIdx.z * MK;
    const int mBlock = blockIdx.y * BLK_M;
    const int nBlock = blockIdx.x * BLK_N;

    const int r0  = tid >> 2;
    const int seg = (tid & 3) * 16;
    const signed char* gAsrc = qa  + batch + (size_t)(mBlock + r0) * MATDIM + seg;
    const signed char* gBsrc = qbt + batch + (size_t)(nBlock + r0) * MATDIM + seg;
    const unsigned lAoff = (unsigned)(unsigned long long)&lA[0][r0 * BK + seg];
    const unsigned lBoff = (unsigned)(unsigned long long)&lB[0][r0 * BK + seg];

    v8i acc[2][4] = {};

    const int aOff = hi * 16;
    const int bOff = hi * 16;

    union AU { v8i v; unsigned long long q[4]; };
    union BU { v8i v; v4i x[2]; };

    const signed char* apBase0 = &lA[0][(wm * 32 +  0 + row) * BK + aOff];
    const signed char* apBase1 = &lA[0][(wm * 32 + 16 + row) * BK + aOff];
    const signed char* bpBase  = &lB[0][(wn * 64 + row) * BK + bOff];

    async_copy_b128(lAoff,            gAsrc, 0);
    async_copy_b128(lAoff + CHUNK1_L, gAsrc, CHUNK1_G);
    async_copy_b128(lBoff,            gBsrc, 0);
    async_copy_b128(lBoff + CHUNK1_L, gBsrc, CHUNK1_G);

#pragma unroll
    for (int kb = 0; kb < KSTEPS; ++kb) {
        const int cur  = kb & 1;
        const int curL = cur * TILE_BYTES;

        async_wait0();
        __syncthreads();

        if (kb + 1 < KSTEPS) {
            const unsigned nb = (unsigned)((cur ^ 1) * TILE_BYTES);
            async_copy_b128(lAoff + nb,            gAsrc, (kb + 1) * BK);
            async_copy_b128(lAoff + nb + CHUNK1_L, gAsrc, (kb + 1) * BK + CHUNK1_G);
            async_copy_b128(lBoff + nb,            gBsrc, (kb + 1) * BK);
            async_copy_b128(lBoff + nb + CHUNK1_L, gBsrc, (kb + 1) * BK + CHUNK1_G);
        }

        AU a[2];
        {
            const signed char* ap0 = apBase0 + curL;
            const signed char* ap1 = apBase1 + curL;
            a[0].q[0] = *(const unsigned long long*)(ap0 +  0);
            a[0].q[1] = *(const unsigned long long*)(ap0 +  8);
            a[0].q[2] = *(const unsigned long long*)(ap0 + 32);
            a[0].q[3] = *(const unsigned long long*)(ap0 + 40);
            a[1].q[0] = *(const unsigned long long*)(ap1 +  0);
            a[1].q[1] = *(const unsigned long long*)(ap1 +  8);
            a[1].q[2] = *(const unsigned long long*)(ap1 + 32);
            a[1].q[3] = *(const unsigned long long*)(ap1 + 40);
        }

#pragma unroll
        for (int j = 0; j < 4; ++j) {
            BU b;
            const signed char* bp = bpBase + curL + j * (16 * BK);
            b.x[0] = *(const v4i*)(bp +  0);
            b.x[1] = *(const v4i*)(bp + 32);
            acc[0][j] = __builtin_amdgcn_wmma_i32_16x16x64_iu8(true, a[0].v, true, b.v, acc[0][j], false, false);
            acc[1][j] = __builtin_amdgcn_wmma_i32_16x16x64_iu8(true, a[1].v, true, b.v, acc[1][j], false, false);
        }
    }

    const int nCol = lane & 15;
    float* outB = out + batch;
#pragma unroll
    for (int i = 0; i < 2; ++i) {
        const int mRow = mBlock + wm * 32 + i * 16 + hi * 8;
#pragma unroll
        for (int j = 0; j < 4; ++j) {
            const int nC = nBlock + wn * 64 + j * 16 + nCol;
#pragma unroll
            for (int r = 0; r < 8; ++r) {
                outB[(size_t)(mRow + r) * MATDIM + nC] = (float)acc[i][j][r] * OSCALE_INV;
            }
        }
    }
}

extern "C" void kernel_launch(void* const* d_in, const int* in_sizes, int n_in,
                              void* d_out, int out_size, void* d_ws, size_t ws_size,
                              hipStream_t stream) {
    const float* in0 = (const float*)d_in[0];
    const float* in1 = (const float*)d_in[1];
    float* out = (float*)d_out;

    const size_t elems = (size_t)B_DIM * MATDIM * MATDIM;
    signed char* qa  = (signed char*)d_ws;
    signed char* qbt = qa + elems;

    quantA_kernel<<<dim3((unsigned)(elems / 4 / 256)), 256, 0, stream>>>(in0, qa);

    quantBT_kernel<<<dim3(MATDIM / 32, MATDIM / 32, B_DIM), 256, 0, stream>>>(in1, qbt);

    gemm_i8_wmma_kernel<<<dim3(MATDIM / BLK_N, MATDIM / BLK_M, B_DIM), 256, 0, stream>>>(qa, qbt, out);
}
